// GaussianNonLocalMeans_3092376453293
// MI455X (gfx1250) — matrix operand order repaired and checked by reading, not compiled, not run on hardware
//
#include <hip/hip_runtime.h>
#include <hip/hip_bf16.h>

// ---------------- problem constants ----------------
#define BATCH 4
#define CC    256      // channels
#define CK    32       // projected channels
#define NTOK  4096     // H*W

typedef __attribute__((ext_vector_type(16))) __bf16 v16bf;
typedef __attribute__((ext_vector_type(8)))  __bf16 v8bf;
typedef __attribute__((ext_vector_type(4)))  __bf16 v4bf;
typedef __attribute__((ext_vector_type(8)))  float  v8f;

static __device__ inline v8f zero8() {
    v8f z;
#pragma unroll
    for (int i = 0; i < 8; ++i) z[i] = 0.0f;
    return z;
}

static __device__ inline v8f wmma_bf16(v16bf a, v16bf b, v8f c) {
    // (neg_a, A, neg_b, B, c_mod, C, reuse_a, reuse_b)
    return __builtin_amdgcn_wmma_f32_16x16x32_bf16(false, a, false, b, (short)0, c, false, false);
}

// Build a 16-element A-fragment from two contiguous 8-element (16B) chunks.
// Order built by the callers (16-bit A 16x32): lane L holds row M=L%16;
//   elems 0..7  -> K = 16*(L/16) + i (callers pass p_hi = p_lo + 8)
//   elems 8..15 -> K = 16*(L/16) + i (contiguous, same as the B loads)
static __device__ inline v16bf frag_a_from2(const __bf16* p_lo, const __bf16* p_hi) {
    v8bf a = *(const v8bf*)p_lo;
    v8bf b = *(const v8bf*)p_hi;
    v16bf r;
#pragma unroll
    for (int i = 0; i < 8; ++i) { r[i] = a[i]; r[i + 8] = b[i]; }
    return r;
}

// ---------------- kernel 1: f32 -> bf16 copy (vector-4) ----------------
__global__ void cvt_bf16_kernel(const float* __restrict__ in, __bf16* __restrict__ out, int n4) {
    int i = blockIdx.x * blockDim.x + threadIdx.x;
    if (i >= n4) return;
    const float4 v = ((const float4*)in)[i];
    v4bf o;
    o[0] = (__bf16)v.x; o[1] = (__bf16)v.y; o[2] = (__bf16)v.z; o[3] = (__bf16)v.w;
    ((v4bf*)out)[i] = o;
}

// ---------------- kernel 1b: transpose [32][256] -> [256][32] (f32) ----------------
// Lets the projection kernel read 32 contiguous uniform floats per channel step
// (wide scalar-cache loads instead of 64 strided s_loads per iteration).
__global__ void wtrans_kernel(const float* __restrict__ w, float* __restrict__ wT) {
    int i = blockIdx.x * 256 + threadIdx.x;   // 0 .. CK*CC-1 (8192)
    int k = i >> 8;        // 0..31
    int c = i & 255;       // 0..255
    wT[c * CK + k] = w[k * CC + c];
}

// ---------------- kernel 2: theta/phi projection ----------------
// Q[b][n][k] = sum_c wT[c][k] * x[b][c][n]   (same for K with w_phi)
// One thread per (b,n): coalesced x reads across n, contiguous uniform w reads.
__global__ __launch_bounds__(256) void proj_kernel(const float* __restrict__ x,
                                                   const float* __restrict__ wtT,
                                                   const float* __restrict__ wpT,
                                                   __bf16* __restrict__ Q,
                                                   __bf16* __restrict__ K) {
    int g = blockIdx.x * 256 + threadIdx.x;     // 0 .. BATCH*NTOK-1
    int b = g >> 12;
    int n = g & (NTOK - 1);
    const float* xb = x + ((size_t)b * CC) * NTOK + n;

    float at[CK], ap[CK];
#pragma unroll
    for (int k = 0; k < CK; ++k) { at[k] = 0.0f; ap[k] = 0.0f; }

    for (int c = 0; c < CC; ++c) {
        float xv = xb[(size_t)c * NTOK];
        const float* wtc = wtT + c * CK;
        const float* wpc = wpT + c * CK;
#pragma unroll
        for (int k = 0; k < CK; ++k) {
            at[k] = fmaf(wtc[k], xv, at[k]);
            ap[k] = fmaf(wpc[k], xv, ap[k]);
        }
    }

    v8bf* qv = (v8bf*)(Q + (size_t)g * CK);
    v8bf* kv = (v8bf*)(K + (size_t)g * CK);
#pragma unroll
    for (int j = 0; j < 4; ++j) {
        v8bf tq, tk;
#pragma unroll
        for (int e = 0; e < 8; ++e) {
            tq[e] = (__bf16)at[j * 8 + e];
            tk[e] = (__bf16)ap[j * 8 + e];
        }
        qv[j] = tq;
        kv[j] = tk;
    }
}

// ---------------- kernel 3: flash attention (shift-free softmax) ----------------
// Per wave: 16 query rows; stream key/value chunks of 32 tokens.
// Scores are 32-term dots of unit-variance projections (|S| <~ 30 << 88), so
// softmax needs no max subtraction in f32: P = exp(S) directly.
// Row sums accumulate in the MATRIX pipe via an extra WMMA against an all-ones
// B fragment (denominator lands in the same C/D layout as the O accumulators),
// so the loop has ZERO cross-lane reductions: no ds_bpermute, no rescaling.
__global__ __launch_bounds__(128) void flash_kernel(const __bf16* __restrict__ Q,
                                                    const __bf16* __restrict__ Kmat,
                                                    const __bf16* __restrict__ X,
                                                    __bf16* __restrict__ Y) {
    __shared__ __bf16 pstage[4][16 * 32];   // per-wave P tile [16 n][32 m]

    const int wave = threadIdx.x >> 5;
    const int lane = threadIdx.x & 31;
    const int half = lane >> 4;
    const int l16  = lane & 15;

    const int b  = blockIdx.x >> 6;                       // 64 blocks per batch
    const int n0 = ((blockIdx.x & 63) << 6) + (wave << 4);

    const __bf16* Qb = Q    + (size_t)b * NTOK * CK;
    const __bf16* Kb = Kmat + (size_t)b * NTOK * CK;
    const __bf16* Xb = X    + (size_t)b * CC * NTOK;

    // Q A-fragment (held for the whole kernel)
    const int qrow = n0 + l16;
    const v16bf qa = frag_a_from2(Qb + (size_t)qrow * CK + 16 * half,
                                  Qb + (size_t)qrow * CK + 8 + 16 * half);

    // all-ones B fragment for the denominator WMMA
    v16bf ones;
#pragma unroll
    for (int i = 0; i < 16; ++i) ones[i] = (__bf16)1.0f;

    v8f acc[16];
#pragma unroll
    for (int t = 0; t < 16; ++t) acc[t] = zero8();
    v8f accl = zero8();

    __bf16* ps = &pstage[wave][0];

    for (int m0 = 0; m0 < NTOK; m0 += 32) {
        // --- scores: two 16x16 tiles, K-dim = Ck = 32 (single wmma each) ---
        v16bf kb0 = *(const v16bf*)(Kb + (size_t)(m0 + l16)      * CK + 16 * half);
        v16bf kb1 = *(const v16bf*)(Kb + (size_t)(m0 + 16 + l16) * CK + 16 * half);
        if (m0 + 32 < NTOK)
            __builtin_prefetch(Kb + (size_t)(m0 + 32 + l16) * CK, 0, 1);

        v8f s0 = wmma_bf16(qa, kb0, zero8());
        v8f s1 = wmma_bf16(qa, kb1, zero8());

        // --- P = exp(S), staged to LDS in [16 n][32 m] A-layout order ---
#pragma unroll
        for (int r = 0; r < 8; ++r) {
            const float p0 = __expf(s0[r]);
            const float p1 = __expf(s1[r]);
            ps[(r + 8 * half) * 32 + l16]      = (__bf16)p0;
            ps[(r + 8 * half) * 32 + 16 + l16] = (__bf16)p1;
        }

        // --- P A-fragment from LDS (same wave; DScnt-ordered, no barrier) ---
        const v16bf pa = frag_a_from2(ps + l16 * 32 + 16 * half,
                                      ps + l16 * 32 + 8 + 16 * half);

        // denominator: accl += P * 1  (row sums, replicated across columns)
        accl = wmma_bf16(pa, ones, accl);

        // --- PV: V = x[b][c][m] is already B-fragment-contiguous along m ---
        const __bf16* Xm = Xb + m0 + 16 * half;
#pragma unroll
        for (int t = 0; t < 16; ++t) {
            v16bf vb = *(const v16bf*)(Xm + (size_t)(t * 16 + l16) * NTOK);
            acc[t] = wmma_bf16(pa, vb, acc[t]);
        }
    }

    // normalize (accl[r] holds the row sum for row r+8*half, lane-local) and emit
    float inv[8];
#pragma unroll
    for (int r = 0; r < 8; ++r) inv[r] = 1.0f / accl[r];
#pragma unroll
    for (int t = 0; t < 16; ++t) {
#pragma unroll
        for (int r = 0; r < 8; ++r) {
            const float v = acc[t][r] * inv[r];
            Y[((size_t)b * NTOK + (n0 + r + 8 * half)) * CC + t * 16 + l16] = (__bf16)v;
        }
    }
}

// ---------------- kernel 4: output 1x1 conv (GEMM) ----------------
// out[b][o][n] = sum_c w_conv[o][c] * Y_attn[b][n][c]
// Per wave: 16 o x 64 n tile, K = 256 in 8 steps of 32. LDS-transposed coalesced stores.
__global__ __launch_bounds__(256) void outconv_kernel(const __bf16* __restrict__ Wb,
                                                      const __bf16* __restrict__ Y,
                                                      float* __restrict__ out) {
    __shared__ float ostage[8][16][66];

    const int wave = threadIdx.x >> 5;
    const int lane = threadIdx.x & 31;
    const int half = lane >> 4;
    const int l16  = lane & 15;

    const int gw  = blockIdx.x * 8 + wave;   // 0 .. 4095
    const int b   = gw >> 10;                // 1024 wave-tasks per batch
    const int rem = gw & 1023;
    const int o0  = (rem >> 6) << 4;         // 16 o-tiles
    const int n0  = (rem & 63) << 6;         // 64 n-tiles of 64

    const __bf16* Yb = Y + (size_t)b * NTOK * CC;

    v8f acc[4];
#pragma unroll
    for (int s = 0; s < 4; ++s) acc[s] = zero8();

#pragma unroll
    for (int kt = 0; kt < 8; ++kt) {
        const int c0 = kt * 32;
        const v16bf wa = frag_a_from2(Wb + (size_t)(o0 + l16) * CC + c0 + 16 * half,
                                      Wb + (size_t)(o0 + l16) * CC + c0 + 8 + 16 * half);
#pragma unroll
        for (int s = 0; s < 4; ++s) {
            v16bf yb = *(const v16bf*)(Yb + (size_t)(n0 + s * 16 + l16) * CC + c0 + 16 * half);
            acc[s] = wmma_bf16(wa, yb, acc[s]);
        }
    }

    // D layout -> LDS [o][n], then coalesced row stores
#pragma unroll
    for (int s = 0; s < 4; ++s)
#pragma unroll
        for (int r = 0; r < 8; ++r)
            ostage[wave][r + 8 * half][s * 16 + l16] = acc[s][r];

    float* outb = out + ((size_t)b * CC + o0) * NTOK + n0;
#pragma unroll
    for (int r = 0; r < 16; ++r) {
        outb[(size_t)r * NTOK + lane]      = ostage[wave][r][lane];
        outb[(size_t)r * NTOK + lane + 32] = ostage[wave][r][lane + 32];
    }
}

// ---------------- host launcher ----------------
extern "C" void kernel_launch(void* const* d_in, const int* in_sizes, int n_in,
                              void* d_out, int out_size, void* d_ws, size_t ws_size,
                              hipStream_t stream) {
    const float* x       = (const float*)d_in[0];   // [4,256,64,64]
    const float* w_theta = (const float*)d_in[1];   // [32,256]
    const float* w_phi   = (const float*)d_in[2];   // [32,256]
    const float* w_conv  = (const float*)d_in[3];   // [256,256]
    float* out = (float*)d_out;                     // [4,256,64,64]

    char* ws = (char*)d_ws;
    __bf16* Qbf = (__bf16*)(ws);                                  // 1 MB
    __bf16* Kbf = (__bf16*)(ws + (1u << 20));                     // 1 MB
    __bf16* Xbf = (__bf16*)(ws + (2u << 20));                     // 8 MB
    __bf16* Wbf = (__bf16*)(ws + (10u << 20));                    // 128 KB
    __bf16* Ybf = (__bf16*)(ws + (10u << 20) + (128u << 10));     // 8 MB
    float*  WtT = (float*)(ws + (19u << 20));                     // 32 KB
    float*  WpT = (float*)(ws + (19u << 20) + (32u << 10));       // 32 KB

    // 1) bf16 conversions + weight transposes
    {
        const int n4 = (BATCH * CC * NTOK) / 4;   // 1,048,576
        cvt_bf16_kernel<<<n4 / 256, 256, 0, stream>>>(x, Xbf, n4);
        const int w4 = (CC * CC) / 4;             // 16,384
        cvt_bf16_kernel<<<w4 / 256, 256, 0, stream>>>(w_conv, Wbf, w4);
        wtrans_kernel<<<(CK * CC) / 256, 256, 0, stream>>>(w_theta, WtT);
        wtrans_kernel<<<(CK * CC) / 256, 256, 0, stream>>>(w_phi, WpT);
    }
    // 2) theta / phi projections -> Q, K (bf16, [b][n][32])
    proj_kernel<<<(BATCH * NTOK) / 256, 256, 0, stream>>>(x, WtT, WpT, Qbf, Kbf);
    // 3) flash attention -> Ybf (bf16, [b][n][c])
    flash_kernel<<<BATCH * (NTOK / 64), 128, 0, stream>>>(Qbf, Kbf, Xbf, Ybf);
    // 4) output 1x1 conv -> out (f32, [b][o][n])
    outconv_kernel<<<(BATCH * 16 * 64) / 8, 256, 0, stream>>>(Wbf, Ybf, out);
}
